// MultiHeadAttention_9955734192512
// MI455X (gfx1250) — compile-verified, matrix operand order checked by reading, not run on hardware
//
#include <hip/hip_runtime.h>
#include <hip/hip_bf16.h>

typedef __attribute__((ext_vector_type(16))) _Float16 v16h;
typedef __attribute__((ext_vector_type(8)))  float    v8f;
typedef __attribute__((ext_vector_type(4)))  int      v4i;

#define D_MODEL 1024
#define SEQ     2048
#define NHEADS  16
#define DK      64

#if defined(__has_builtin)
#if __has_builtin(__builtin_amdgcn_global_load_async_to_lds_b128)
#define HAVE_ASYNC_LDS 1
#endif
#endif

#define AS1 __attribute__((address_space(1)))
#define AS3 __attribute__((address_space(3)))

union FragH {
    v16h v;
    uint4 q[2];
    _Float16 h[16];
};

// D = A(16x32 f16) * B(32x16 f16) + C(16x16 f32)
__device__ inline v8f wmma32(v16h a, v16h b, v8f c) {
    return __builtin_amdgcn_wmma_f32_16x16x32_f16(false, a, false, b, (short)0, c,
                                                  false, false);
}

// Canonical 16x32 fragment load: "row" dim -> lane%16, K contiguous in memory.
// (ISA 7.12.2: lane<16 holds K {0..7,16..23}, lane>=16 holds K {8..15,24..31}.)
template <typename P>
__device__ inline v16h load_frag_f16(P p, int ld) {
    int lane = threadIdx.x & 31;
    int r  = lane & 15;
    int kg = (lane >> 4) << 3;   // 0 or 8
    auto row = p + (size_t)r * ld;
    FragH f;
    f.q[0] = *(const uint4*)(row + kg);
    f.q[1] = *(const uint4*)(row + 16 + kg);
    return f.v;
}

// Same pattern, fp32 source, convert on load.
__device__ inline v16h load_frag_f32(const float* __restrict__ p, int ld) {
    int lane = threadIdx.x & 31;
    int r  = lane & 15;
    int kg = (lane >> 4) << 3;
    const float* row = p + (size_t)r * ld;
    FragH f;
#pragma unroll
    for (int i = 0; i < 8; ++i) f.h[i] = (_Float16)row[kg + i];
#pragma unroll
    for (int i = 0; i < 8; ++i) f.h[8 + i] = (_Float16)row[16 + kg + i];
    return f.v;
}

// 16-byte global->LDS copy chunk, async if available.
__device__ inline void copy16_async(const _Float16* src, _Float16* dst) {
#ifdef HAVE_ASYNC_LDS
    __builtin_amdgcn_global_load_async_to_lds_b128(
        (AS1 v4i*)(uintptr_t)src, (AS3 v4i*)(uintptr_t)dst, 0, 0);
#else
    *(uint4*)dst = *(const uint4*)src;
#endif
}

// Wait until this wave's outstanding async-LDS loads are <= N.
// Async loads complete in order, so N = (#loads of the in-flight next tile)
// guarantees the previous tile is fully resident.
template <int N>
__device__ inline void async_wait() {
#ifdef HAVE_ASYNC_LDS
    asm volatile("s_wait_asynccnt %0" :: "i"(N) : "memory");
#endif
}

// ---------------------------------------------------------------------------
// Kernel 1: convert the 4 weight matrices fp32 -> f16 into workspace.
// ---------------------------------------------------------------------------
__global__ void cvt_weights_kernel(const float* __restrict__ Wq,
                                   const float* __restrict__ Wk,
                                   const float* __restrict__ Wv,
                                   const float* __restrict__ Wo,
                                   _Float16* __restrict__ out) {
    const size_t N = (size_t)D_MODEL * D_MODEL;
    size_t i = (size_t)blockIdx.x * blockDim.x + threadIdx.x;
    const float* src;
    size_t j;
    if (i < N)            { src = Wq; j = i; }
    else if (i < 2 * N)   { src = Wk; j = i - N; }
    else if (i < 3 * N)   { src = Wv; j = i - 2 * N; }
    else                  { src = Wo; j = i - 3 * N; }
    out[i] = (_Float16)src[j];
}

// ---------------------------------------------------------------------------
// Kernel 2: projection GEMM  P = X @ W^T + b. Wave computes 16x64 strip.
// MODE 0: head-split [B,H,S,DK] (q,k).  MODE 1: transposed [B,H,DK,S] (V^T).
// ---------------------------------------------------------------------------
template <int MODE>
__global__ __launch_bounds__(128, 1)
void proj_kernel(const float* __restrict__ X,
                 const _Float16* __restrict__ W,
                 const float* __restrict__ bias,
                 _Float16* __restrict__ out) {
    const int D = D_MODEL;
    int wave = (int)((blockIdx.x * blockDim.x + threadIdx.x) >> 5);
    int lane = threadIdx.x & 31;
    int mt  = wave >> 4;
    int n64 = wave & 15;
    int m0  = mt * 16;

    v8f acc[4] = {};
    for (int k0 = 0; k0 < D; k0 += 32) {
        if (k0 + 32 < D) {
            // warm next K-slice (lowers to global_prefetch_b8)
            __builtin_prefetch(X + (size_t)(m0 + (threadIdx.x & 15)) * D + k0 + 32, 0, 1);
            __builtin_prefetch(W + (size_t)(n64 * 64 + (threadIdx.x & 63)) * D + k0 + 32, 0, 1);
        }
        v16h a = load_frag_f32(X + (size_t)m0 * D + k0, D);
#pragma unroll
        for (int nt = 0; nt < 4; ++nt) {
            v16h b = load_frag_f16(W + (size_t)(n64 * 64 + nt * 16) * D + k0, D);
            acc[nt] = wmma32(a, b, acc[nt]);
        }
    }

    int col  = lane & 15;
    int half = lane >> 4;
#pragma unroll
    for (int nt = 0; nt < 4; ++nt) {
        int n = n64 * 64 + nt * 16 + col;
        float bv = bias[n];
        int h = n >> 6;
        int d = n & 63;
#pragma unroll
        for (int r = 0; r < 8; ++r) {
            int m = m0 + half * 8 + r;
            int b_ = m / SEQ;
            int s  = m % SEQ;
            float val = acc[nt][r] + bv;
            size_t idx;
            if (MODE == 0)
                idx = (((size_t)(b_ * NHEADS + h) * SEQ) + s) * DK + d;
            else
                idx = (((size_t)(b_ * NHEADS + h) * DK) + d) * SEQ + s;
            out[idx] = (_Float16)val;
        }
    }
}

// ---------------------------------------------------------------------------
// Kernel 3: attention, transposed-score formulation.
//   S^T = K_tile @ Q^T  -> C layout: lane = query, reg = key
//   => softmax state is lane-private; exp(S^T) regs ARE the B frag of V^T@P^T
// 4 waves/block share (b,h); K/V tiles double-buffered in LDS via async loads
// (issue next tile, wait ASYNCcnt<=4, compute current while next streams in).
// ---------------------------------------------------------------------------
__global__ __launch_bounds__(128, 1)
void attn_kernel(const _Float16* __restrict__ qh,
                 const _Float16* __restrict__ kh,
                 const _Float16* __restrict__ vT,
                 _Float16* __restrict__ ctx) {
    __shared__ alignas(16) _Float16 kt[2][32 * DK];   // [buf][key][d]   2x4 KB
    __shared__ alignas(16) _Float16 vt[2][DK * 32];   // [buf][d][key]   2x4 KB

    int tid  = threadIdx.x;
    int wv   = tid >> 5;
    int lane = tid & 31;
    int bh   = blockIdx.x >> 5;              // 0..31 = b*16+h
    int qt   = (blockIdx.x & 31) * 4 + wv;   // 0..127

    const _Float16* qbase = qh + ((size_t)bh * SEQ + qt * 16) * DK;
    const _Float16* kbase = kh + (size_t)bh * SEQ * DK;
    const _Float16* vbase = vT + (size_t)bh * DK * SEQ;

    // Issue 4 async 16B chunks per thread: K tile (32x64) + V^T tile (64x32).
    auto stage = [&](int buf, int kb) {
#pragma unroll
        for (int i = 0; i < 2; ++i) {
            int t = tid * 2 + i;                       // 0..255
            copy16_async(kbase + (size_t)(kb + (t >> 3)) * DK + (t & 7) * 8,
                         &kt[buf][t * 8]);
            copy16_async(vbase + (size_t)(t >> 2) * SEQ + kb + (t & 3) * 8,
                         &vt[buf][t * 8]);
        }
    };

    // Q as B fragments (lane = query, d contiguous), loaded once.
    v16h qB0 = load_frag_f16(qbase, DK);        // d 0..31
    v16h qB1 = load_frag_f16(qbase + 32, DK);   // d 32..63

    v8f acc[4] = {};          // out^T: reg = d, lane = query
    float rsum = 0.0f;        // this lane-half's partial sum for its query
    const float scale = 0.125f;
    const int NIT = SEQ / 32;

    stage(0, 0);
    for (int it = 0; it < NIT; ++it) {
        int buf = it & 1;
        if (it + 1 < NIT) {
            stage(buf ^ 1, (it + 1) * 32);
            async_wait<4>();      // the 4 just-issued may be in flight;
                                  // in-order completion => current tile done
        } else {
            async_wait<0>();
        }
        __syncthreads();          // current tile visible to all 4 waves

        const _Float16* ktb = kt[buf];
        const _Float16* vtb = vt[buf];

        // ---- S^T tiles: M = key (2 tiles of 16), N = query ----
        v8f St0 = {}, St1 = {};
        {
            v16h kA0 = load_frag_f16(ktb, DK);
            v16h kA1 = load_frag_f16(ktb + 32, DK);
            St0 = wmma32(kA0, qB0, St0);
            St0 = wmma32(kA1, qB1, St0);
            v16h kA2 = load_frag_f16(ktb + 16 * DK, DK);
            v16h kA3 = load_frag_f16(ktb + 16 * DK + 32, DK);
            St1 = wmma32(kA2, qB0, St1);
            St1 = wmma32(kA3, qB1, St1);
        }

        // ---- exp + pack: C layout of S^T == B layout of P^T ----
        FragH pb;
        float ls = 0.0f;
#pragma unroll
        for (int r = 0; r < 8; ++r) {
            float p0 = __expf(St0[r] * scale);
            float p1 = __expf(St1[r] * scale);
            pb.h[r]     = (_Float16)p0;
            pb.h[8 + r] = (_Float16)p1;
            ls += p0 + p1;
        }
        rsum += ls;

        // ---- out^T += V^T_tile(16d x 32keys) @ P^T(32keys x 16q) ----
#pragma unroll
        for (int dt = 0; dt < 4; ++dt) {
            v16h vA = load_frag_f16(vtb + dt * 16 * 32, 32);
            acc[dt] = wmma32(vA, pb.v, acc[dt]);
        }
        __syncthreads();          // all waves done reading buf before it is
                                  // overwritten by stage() next iteration
    }

    // ---- merge the two lane-halves' key partitions, normalize, store ----
    float rtot = rsum + __shfl_xor(rsum, 16, 32);
    float inv  = 1.0f / rtot;

    int b_   = bh >> 4;
    int h    = bh & 15;
    int q    = qt * 16 + (lane & 15);
    int half = lane >> 4;
    _Float16* dst = ctx + ((size_t)b_ * SEQ + q) * D_MODEL + h * DK;
#pragma unroll
    for (int dt = 0; dt < 4; ++dt)
#pragma unroll
        for (int r = 0; r < 8; ++r)
            dst[dt * 16 + half * 8 + r] = (_Float16)(acc[dt][r] * inv);
}

// ---------------------------------------------------------------------------
// Kernel 4: output projection  out = ctx @ Wo^T + bo   (fp32 out).
// ---------------------------------------------------------------------------
__global__ __launch_bounds__(128, 1)
void oproj_kernel(const _Float16* __restrict__ A,
                  const _Float16* __restrict__ W,
                  const float* __restrict__ bias,
                  float* __restrict__ out) {
    const int D = D_MODEL;
    int wave = (int)((blockIdx.x * blockDim.x + threadIdx.x) >> 5);
    int lane = threadIdx.x & 31;
    int mt  = wave >> 4;
    int n64 = wave & 15;
    int m0  = mt * 16;

    v8f acc[4] = {};
    for (int k0 = 0; k0 < D; k0 += 32) {
        if (k0 + 32 < D) {
            __builtin_prefetch(A + (size_t)(m0 + (threadIdx.x & 15)) * D + k0 + 32, 0, 1);
            __builtin_prefetch(W + (size_t)(n64 * 64 + (threadIdx.x & 63)) * D + k0 + 32, 0, 1);
        }
        v16h a = load_frag_f16(A + (size_t)m0 * D + k0, D);
#pragma unroll
        for (int nt = 0; nt < 4; ++nt) {
            v16h b = load_frag_f16(W + (size_t)(n64 * 64 + nt * 16) * D + k0, D);
            acc[nt] = wmma32(a, b, acc[nt]);
        }
    }

    int col  = lane & 15;
    int half = lane >> 4;
#pragma unroll
    for (int nt = 0; nt < 4; ++nt) {
        int n = n64 * 64 + nt * 16 + col;
        float bv = bias[n];
#pragma unroll
        for (int r = 0; r < 8; ++r) {
            int m = m0 + half * 8 + r;
            out[(size_t)m * D + n] = acc[nt][r] + bv;
        }
    }
}

// ---------------------------------------------------------------------------
// Launcher
// ---------------------------------------------------------------------------
extern "C" void kernel_launch(void* const* d_in, const int* in_sizes, int n_in,
                              void* d_out, int out_size, void* d_ws, size_t ws_size,
                              hipStream_t stream) {
    const float* Q  = (const float*)d_in[0];
    const float* K  = (const float*)d_in[1];
    const float* V  = (const float*)d_in[2];
    const float* Wq = (const float*)d_in[3];
    const float* bq = (const float*)d_in[4];
    const float* Wk = (const float*)d_in[5];
    const float* bk = (const float*)d_in[6];
    const float* Wv = (const float*)d_in[7];
    const float* bv = (const float*)d_in[8];
    const float* Wo = (const float*)d_in[9];
    const float* bo = (const float*)d_in[10];
    float* out = (float*)d_out;

    _Float16* ws = (_Float16*)d_ws;
    const size_t N = (size_t)D_MODEL * D_MODEL;  // 1,048,576
    _Float16* Woh = ws + 3 * N;    // [3N, 4N)
    _Float16* qh  = ws + 4 * N;    // [4N,  8N)  [B,H,S,DK]
    _Float16* kh  = ws + 8 * N;    // [8N, 12N)  [B,H,S,DK]
    _Float16* vT  = ws + 12 * N;   // [12N,16N)  [B,H,DK,S]
    _Float16* ctx = ws + 16 * N;   // [16N,20N)  [B,S,D_MODEL]

    // 1) weights fp32 -> f16
    cvt_weights_kernel<<<16384, 256, 0, stream>>>(Wq, Wk, Wv, Wo, ws);

    // 2) projections (256 m-tiles x 16 n64-tiles = 4096 waves)
    proj_kernel<0><<<1024, 128, 0, stream>>>(Q, ws, bq, qh);
    proj_kernel<0><<<1024, 128, 0, stream>>>(K, ws + N, bk, kh);
    proj_kernel<1><<<1024, 128, 0, stream>>>(V, ws + 2 * N, bv, vT);

    // 3) attention (32 bh x 32 qt-groups = 1024 blocks, 4 waves each)
    attn_kernel<<<1024, 128, 0, stream>>>(qh, kh, vT, ctx);

    // 4) output projection
    oproj_kernel<<<1024, 128, 0, stream>>>(ctx, Woh, bo, out);
}
